// LIFNeuron_56693568307788
// MI455X (gfx1250) — compile-verified, matrix operand order checked by reading, not run on hardware
//
#include <hip/hip_runtime.h>

typedef __attribute__((ext_vector_type(16))) _Float16 v16h;
typedef __attribute__((ext_vector_type(8)))  _Float16 v8h;
typedef __attribute__((ext_vector_type(8)))  float    v8f;
typedef __attribute__((ext_vector_type(4)))  float    v4f;

#define DIM 64
#define WAVES 4            // waves per block (128 threads)
#define STAGSTR 80         // f16 elems per staged row (64 + pad, keeps 16B align)
#define HSTEP 0.125f       // 1/N_STEPS
#define THRESH 0.3f
#define L2E 1.44269504088896f

struct Tile { v8f t[4]; };  // 16x64 f32 tile in WMMA C/D layout (4 N-tiles of v8f)

__device__ __forceinline__ v16h cat16(v8h lo, v8h hi) {
  v16h r;
#pragma unroll
  for (int i = 0; i < 8; ++i) { r[i] = lo[i]; r[i + 8] = hi[i]; }
  return r;
}

// branchless exp(x) through the native v_exp_f32 (2^y) unit; caller guarantees range
__device__ __forceinline__ float fast_exp(float x) {
  return __builtin_amdgcn_exp2f(x * L2E);
}

__global__ __launch_bounds__(WAVES * 32)
void lif_dopri5_wmma_kernel(const float* __restrict__ xin,
                            const float* __restrict__ Wo,
                            const float* __restrict__ Wg,
                            const float* __restrict__ bgp,
                            float* __restrict__ out,
                            int batch) {
  // 16 B-layout weight fragments: f = m*8 + kc*4 + n  (m: 0=W,1=Wg; kc: K-chunk; n: N-tile)
  __shared__ __align__(32) _Float16 wfrag[16][512];                 // 16 KB
  __shared__ __align__(32) _Float16 stag[WAVES][16][STAGSTR];       // 10 KB, per-wave A staging
  __shared__ __align__(16) v4f      kbuf[WAVES][5][4][2][32];       // 80 KB, lane-private k spill

  const int tid  = threadIdx.x;
  const int w    = tid >> 5;
  const int lane = tid & 31;
  const int lo   = lane & 15;   // C/D: column within N-tile;  A: row M
  const int hi   = lane >> 4;   // C/D: row-half select;       A: K-half select

  // ---- one-time: convert W / Wg into f16 WMMA B-layout fragments in LDS ----
  const float* wsrc[2] = { Wo, Wg };
#pragma unroll
  for (int fi = 0; fi < 4; ++fi) {
    const int f  = w * 4 + fi;                 // 4 frags per wave, 16 total
    const int m  = f >> 3;
    const int kc = (f >> 2) & 1;
    const int n  = f & 3;
#pragma unroll
    for (int p = 0; p < 16; ++p) {
      const int K = 32 * kc + (p >> 3) * 16 + hi * 8 + (p & 7);
      const int N = 16 * n + lo;
      wfrag[f][lane * 16 + p] = (_Float16)wsrc[m][K * DIM + N];
    }
  }
  float bgv[4];
#pragma unroll
  for (int n = 0; n < 4; ++n) bgv[n] = bgp[16 * n + lo];
  __syncthreads();

  const int row0 = (blockIdx.x * WAVES + w) * 16;

  // ---- load 16x64 x-tile into C/D-layout registers ----
  Tile x;
#pragma unroll
  for (int n = 0; n < 4; ++n)
#pragma unroll
    for (int r = 0; r < 8; ++r) {
      const int row = row0 + r + 8 * hi;
      x.t[n][r] = (row < batch) ? xin[row * DIM + 16 * n + lo] : 0.0f;
    }

  // dx = gelu(xi@W) / (1 + sigmoid(xi@Wg + bg)) via 16 WMMAs per stage
  auto dynamics = [&](const Tile& xi, Tile& k) {
    asm volatile("s_wait_dscnt 0" ::: "memory");
#pragma unroll
    for (int n = 0; n < 4; ++n)
#pragma unroll
      for (int r = 0; r < 8; ++r)
        stag[w][r + 8 * hi][16 * n + lo] = (_Float16)xi.t[n][r];
    asm volatile("s_wait_dscnt 0" ::: "memory");

    // Build A fragments (16x32 f16 each): lane row = lo, K halves per A-layout
    const _Float16* rp = &stag[w][lo][0];
    const v16h a0 = cat16(*(const v8h*)(rp +      hi * 8), *(const v8h*)(rp + 16 + hi * 8));
    const v16h a1 = cat16(*(const v8h*)(rp + 32 + hi * 8), *(const v8h*)(rp + 48 + hi * 8));

#pragma unroll
    for (int n = 0; n < 4; ++n) {
      v8f u = {}, g = {};
      const v16h b0 = *(const v16h*)&wfrag[ 0 + n][lane * 16];   // W,  K0
      u = __builtin_amdgcn_wmma_f32_16x16x32_f16(false, a0, false, b0, (short)0, u, false, false);
      const v16h b1 = *(const v16h*)&wfrag[ 4 + n][lane * 16];   // W,  K1
      u = __builtin_amdgcn_wmma_f32_16x16x32_f16(false, a1, false, b1, (short)0, u, false, false);
      const v16h b2 = *(const v16h*)&wfrag[ 8 + n][lane * 16];   // Wg, K0
      g = __builtin_amdgcn_wmma_f32_16x16x32_f16(false, a0, false, b2, (short)0, g, false, false);
      const v16h b3 = *(const v16h*)&wfrag[12 + n][lane * 16];   // Wg, K1
      g = __builtin_amdgcn_wmma_f32_16x16x32_f16(false, a1, false, b3, (short)0, g, false, false);
#pragma unroll
      for (int r = 0; r < 8; ++r) {
        // time factor: tf = (1+e^{-g})/(2+e^{-g}), branchless + saturating
        const float gv = fminf(fmaxf(g[r] + bgv[n], -30.0f), 30.0f);
        const float e  = fast_exp(-gv);
        const float tf = (1.0f + e) * __builtin_amdgcn_rcpf(2.0f + e);
        // gelu(x) ~= 0.5*x*(1 + tanh(0.79788456*(x + 0.044715*x^3))), branchless
        const float uu = u[r];
        float inner = uu * __builtin_fmaf(uu * uu, 0.0356774081f, 0.7978845608f);
        inner = fminf(fmaxf(inner, -15.0f), 15.0f);
        const float tq = __builtin_amdgcn_exp2f(inner * (-2.0f * L2E)); // e^{-2*inner}
        const float th = (1.0f - tq) * __builtin_amdgcn_rcpf(1.0f + tq);
        const float gl = 0.5f * uu * (1.0f + th);
        k.t[n][r] = gl * tf;
      }
    }
  };

  auto kstore = [&](int j, const Tile& k) {
#pragma unroll
    for (int n = 0; n < 4; ++n) {
      kbuf[w][j][n][0][lane] = v4f{k.t[n][0], k.t[n][1], k.t[n][2], k.t[n][3]};
      kbuf[w][j][n][1][lane] = v4f{k.t[n][4], k.t[n][5], k.t[n][6], k.t[n][7]};
    }
  };
  auto kload = [&](int j, int n) -> v8f {
    const v4f p0 = kbuf[w][j][n][0][lane];
    const v4f p1 = kbuf[w][j][n][1][lane];
    v8f r;
#pragma unroll
    for (int i = 0; i < 4; ++i) { r[i] = p0[i]; r[i + 4] = p1[i]; }
    return r;
  };

  Tile xi, k;
#pragma unroll 1
  for (int step = 0; step < 8; ++step) {
    // stage 1: k1 = f(x)
    dynamics(x, k); kstore(0, k);
    // stage 2
#pragma unroll
    for (int n = 0; n < 4; ++n)
      xi.t[n] = x.t[n] + (HSTEP * 0.2f) * k.t[n];
    dynamics(xi, k); kstore(1, k);
    // stage 3
#pragma unroll
    for (int n = 0; n < 4; ++n)
      xi.t[n] = x.t[n] + (HSTEP * 0.075f) * kload(0, n)
                       + (HSTEP * 0.225f) * k.t[n];
    dynamics(xi, k); kstore(2, k);
    // stage 4
#pragma unroll
    for (int n = 0; n < 4; ++n)
      xi.t[n] = x.t[n] + (HSTEP * (44.0f/45.0f))  * kload(0, n)
                       + (HSTEP * (-56.0f/15.0f)) * kload(1, n)
                       + (HSTEP * (32.0f/9.0f))   * k.t[n];
    dynamics(xi, k); kstore(3, k);
    // stage 5
#pragma unroll
    for (int n = 0; n < 4; ++n)
      xi.t[n] = x.t[n] + (HSTEP * (19372.0f/6561.0f))  * kload(0, n)
                       + (HSTEP * (-25360.0f/2187.0f)) * kload(1, n)
                       + (HSTEP * (64448.0f/6561.0f))  * kload(2, n)
                       + (HSTEP * (-212.0f/729.0f))    * k.t[n];
    dynamics(xi, k); kstore(4, k);
    // stage 6
#pragma unroll
    for (int n = 0; n < 4; ++n)
      xi.t[n] = x.t[n] + (HSTEP * (9017.0f/3168.0f))   * kload(0, n)
                       + (HSTEP * (-355.0f/33.0f))     * kload(1, n)
                       + (HSTEP * (46732.0f/5247.0f))  * kload(2, n)
                       + (HSTEP * (49.0f/176.0f))      * kload(3, n)
                       + (HSTEP * (-5103.0f/18656.0f)) * k.t[n];
    dynamics(xi, k);   // k = k6 (stage 7 is dead: b7 == 0)
    // 5th-order update: x += h*(b1 k1 + b3 k3 + b4 k4 + b5 k5 + b6 k6)
#pragma unroll
    for (int n = 0; n < 4; ++n)
      x.t[n] = x.t[n] + (HSTEP * (35.0f/384.0f))      * kload(0, n)
                      + (HSTEP * (500.0f/1113.0f))    * kload(2, n)
                      + (HSTEP * (125.0f/192.0f))     * kload(3, n)
                      + (HSTEP * (-2187.0f/6784.0f))  * kload(4, n)
                      + (HSTEP * (11.0f/84.0f))       * k.t[n];
  }

  // spike = (x_new - THRESHOLD) > 0
#pragma unroll
  for (int n = 0; n < 4; ++n)
#pragma unroll
    for (int r = 0; r < 8; ++r) {
      const int row = row0 + r + 8 * hi;
      if (row < batch)
        out[row * DIM + 16 * n + lo] = (x.t[n][r] - THRESH > 0.0f) ? 1.0f : 0.0f;
    }
}

extern "C" void kernel_launch(void* const* d_in, const int* in_sizes, int n_in,
                              void* d_out, int out_size, void* d_ws, size_t ws_size,
                              hipStream_t stream) {
  (void)n_in; (void)out_size; (void)d_ws; (void)ws_size;
  const float* x  = (const float*)d_in[0];
  const float* Wo = (const float*)d_in[1];
  const float* Wg = (const float*)d_in[2];
  const float* bg = (const float*)d_in[3];
  float* out = (float*)d_out;

  const int batch = in_sizes[0] / DIM;                // 1048576
  const int rows_per_block = WAVES * 16;              // 64
  const int blocks = (batch + rows_per_block - 1) / rows_per_block;

  lif_dopri5_wmma_kernel<<<blocks, WAVES * 32, 0, stream>>>(x, Wo, Wg, bg, out, batch);
}
